// complex_fuse_base_2302102471108
// MI455X (gfx1250) — matrix operand order repaired and checked by reading, not compiled, not run on hardware
//
#include <hip/hip_runtime.h>
#include <hip/hip_bf16.h>

// ---------------- problem constants (match reference) ----------------
constexpr int Pn    = 20000;      // proteins
constexpr int Dn    = 20000;      // drugs
constexpr int SEQL  = 1024;
constexpr int OUTF  = 256;        // H == OUT == 256
constexpr int EPPI  = 640000;
constexpr int EDDI  = 640000;
constexpr int EDTI  = 400000;
constexpr int NPAIR = 100000;

typedef __attribute__((ext_vector_type(16))) __bf16 v16bf;
typedef __attribute__((ext_vector_type(8)))  float  v8f;

#define ACT_NONE 0
#define ACT_RELU 1
#define ACT_ELU  2

// =====================================================================
// GEMM: C[M,N] = act(A[M,K] @ W[N,K]^T + bias)
// One wave32 per 32x32 output macro-tile = 2x2 WMMA tiles: 4 f32
// accumulators, 4 v_wmma_f32_16x16x32_bf16 per K-step sharing two
// A-fragments and two B-fragments (16 loads per WMMA instead of 32).
// Fragment layouts per ISA 16-bit A 16x32 / dense B 32x16 tables.
// M,N multiples of 32; K multiple of 32 (true for all call sites).
// =====================================================================
__device__ __forceinline__ float gemm_act(float v, int act)
{
    if (act == ACT_RELU)     return fmaxf(v, 0.0f);
    if (act == ACT_ELU)      return (v > 0.0f) ? v : (__expf(v) - 1.0f);
    return v;
}

// A 16x32 fragment: lane group hi holds K {hi*8+e} and {16+hi*8+e}, row = lane&15
__device__ __forceinline__ v16bf load_fragA(const float* __restrict__ row, int k0, int hi)
{
    const float4 f0 = *(const float4*)(row + k0 + hi * 8);
    const float4 f1 = *(const float4*)(row + k0 + hi * 8 + 4);
    const float4 f2 = *(const float4*)(row + k0 + 16 + hi * 8);
    const float4 f3 = *(const float4*)(row + k0 + 16 + hi * 8 + 4);
    v16bf a;
    a[0]=(__bf16)f0.x;  a[1]=(__bf16)f0.y;  a[2]=(__bf16)f0.z;  a[3]=(__bf16)f0.w;
    a[4]=(__bf16)f1.x;  a[5]=(__bf16)f1.y;  a[6]=(__bf16)f1.z;  a[7]=(__bf16)f1.w;
    a[8]=(__bf16)f2.x;  a[9]=(__bf16)f2.y;  a[10]=(__bf16)f2.z; a[11]=(__bf16)f2.w;
    a[12]=(__bf16)f3.x; a[13]=(__bf16)f3.y; a[14]=(__bf16)f3.z; a[15]=(__bf16)f3.w;
    return a;
}

// B 32x16 fragment: lane group hi holds K {hi*8+e} and {16+hi*8+e}, column = lane&15
__device__ __forceinline__ v16bf load_fragB(const float* __restrict__ row, int k0, int hi)
{
    const float4 f0 = *(const float4*)(row + k0 + hi * 8);
    const float4 f1 = *(const float4*)(row + k0 + hi * 8 + 4);
    const float4 f2 = *(const float4*)(row + k0 + 16 + hi * 8);
    const float4 f3 = *(const float4*)(row + k0 + 16 + hi * 8 + 4);
    v16bf b;
    b[0]=(__bf16)f0.x;  b[1]=(__bf16)f0.y;  b[2]=(__bf16)f0.z;  b[3]=(__bf16)f0.w;
    b[4]=(__bf16)f1.x;  b[5]=(__bf16)f1.y;  b[6]=(__bf16)f1.z;  b[7]=(__bf16)f1.w;
    b[8]=(__bf16)f2.x;  b[9]=(__bf16)f2.y;  b[10]=(__bf16)f2.z; b[11]=(__bf16)f2.w;
    b[12]=(__bf16)f3.x; b[13]=(__bf16)f3.y; b[14]=(__bf16)f3.z; b[15]=(__bf16)f3.w;
    return b;
}

__global__ void gemm_bf16_wmma(const float* __restrict__ A,
                               const float* __restrict__ W,
                               const float* __restrict__ bias,
                               float* __restrict__ C,
                               int M, int N, int K, int act)
{
    const int wave = threadIdx.x >> 5;
    const int lane = threadIdx.x & 31;
    const int tilesN = N >> 5;                    // 32-wide macro columns
    const int numTiles = (M >> 5) * tilesN;
    const long tile = (long)blockIdx.x * 8 + wave;
    if (tile >= numTiles) return;                 // wave-uniform branch

    const int tm = (int)(tile / tilesN);
    const int tn = (int)(tile % tilesN);
    const int hi = lane >> 4;                     // lane group 0/1
    const int la = lane & 15;

    const float* __restrict__ a0row = A + (long)(tm * 32 + la) * K;
    const float* __restrict__ a1row = a0row + 16L * K;
    const float* __restrict__ b0row = W + (long)(tn * 32 + la) * K;
    const float* __restrict__ b1row = b0row + 16L * K;

    v8f c00 = {}, c01 = {}, c10 = {}, c11 = {};
    for (int k0 = 0; k0 < K; k0 += 32) {
        __builtin_prefetch(a0row + k0 + 64, 0, 1); // global_prefetch_b8
        __builtin_prefetch(a1row + k0 + 64, 0, 1);
        __builtin_prefetch(b0row + k0 + 64, 0, 1);
        __builtin_prefetch(b1row + k0 + 64, 0, 1);
        const v16bf a0 = load_fragA(a0row, k0, hi);
        const v16bf a1 = load_fragA(a1row, k0, hi);
        const v16bf b0 = load_fragB(b0row, k0, hi);
        const v16bf b1 = load_fragB(b1row, k0, hi);
        c00 = __builtin_amdgcn_wmma_f32_16x16x32_bf16(false, a0, false, b0, (short)0, c00, false, false);
        c01 = __builtin_amdgcn_wmma_f32_16x16x32_bf16(false, a0, false, b1, (short)0, c01, false, false);
        c10 = __builtin_amdgcn_wmma_f32_16x16x32_bf16(false, a1, false, b0, (short)0, c10, false, false);
        c11 = __builtin_amdgcn_wmma_f32_16x16x32_bf16(false, a1, false, b1, (short)0, c11, false, false);
    }

    // C/D layout: VGPR v -> row 8*hi + v, col = lane&15
    const int col0 = tn * 32 + la;
    const int col1 = col0 + 16;
    const float bv0 = bias ? bias[col0] : 0.0f;
    const float bv1 = bias ? bias[col1] : 0.0f;
#pragma unroll
    for (int v = 0; v < 8; ++v) {
        const int row0 = tm * 32 + hi * 8 + v;
        const int row1 = row0 + 16;
        C[(long)row0 * N + col0] = gemm_act(c00[v] + bv0, act);
        C[(long)row0 * N + col1] = gemm_act(c01[v] + bv1, act);
        C[(long)row1 * N + col0] = gemm_act(c10[v] + bv0, act);
        C[(long)row1 * N + col1] = gemm_act(c11[v] + bv1, act);
    }
}

// ---------------------------- utility kernels ------------------------
__global__ void fill_kernel(float* __restrict__ p, float v, long n)
{
    long i = (long)blockIdx.x * blockDim.x + threadIdx.x;
    const long st = (long)gridDim.x * blockDim.x;
    for (; i < n; i += st) p[i] = v;
}

__global__ void count_kernel(const int* __restrict__ idx, int E, float* __restrict__ cnt)
{
    long i = (long)blockIdx.x * blockDim.x + threadIdx.x;
    const long st = (long)gridDim.x * blockDim.x;
    for (; i < E; i += st) atomicAdd(&cnt[idx[i]], 1.0f);
}

__global__ void rsqrt_kernel(float* __restrict__ a, int n)
{
    long i = (long)blockIdx.x * blockDim.x + threadIdx.x;
    if (i < n) a[i] = rsqrtf(a[i]);     // deg >= 1 always (self loops)
}

// out[r,f] = bias[f] + dinv[r]^2 * h[r,f]   (self-loop term of GCN)
__global__ void gcn_init_kernel(const float* __restrict__ h,
                                const float* __restrict__ dinv,
                                const float* __restrict__ bias,
                                float* __restrict__ out, long rows)
{
    long i = (long)blockIdx.x * blockDim.x + threadIdx.x;
    const long st = (long)gridDim.x * blockDim.x;
    const long n = rows * OUTF;
    for (; i < n; i += st) {
        const long r = i >> 8;
        const int  f = (int)(i & 255);
        const float di = dinv[r];
        out[i] = bias[f] + di * di * h[i];
    }
}

// out[dst] += dinv[src]*dinv[dst]*h[src]   (one edge per block, 256 feats)
__global__ void gcn_edge_kernel(const float* __restrict__ h,
                                const int* __restrict__ src,
                                const int* __restrict__ dst,
                                const float* __restrict__ dinv,
                                float* __restrict__ out, int E)
{
    const int e = blockIdx.x;
    if (e >= E) return;
    const int s = src[e], d = dst[e];
    const float nrm = dinv[s] * dinv[d];
    const int f = threadIdx.x;
    atomicAdd(&out[(long)d * OUTF + f], nrm * h[(long)s * OUTF + f]);
}

// out[dst_idx[e]] += t[row_idx[e]]
__global__ void smean_scatter_kernel(const float* __restrict__ t,
                                     const int* __restrict__ row_idx,
                                     const int* __restrict__ dst_idx,
                                     float* __restrict__ out, int E)
{
    const int e = blockIdx.x;
    if (e >= E) return;
    const int r = row_idx[e], d = dst_idx[e];
    const int f = threadIdx.x;
    atomicAdd(&out[(long)d * OUTF + f], t[(long)r * OUTF + f]);
}

__global__ void smean_div_kernel(float* __restrict__ out,
                                 const float* __restrict__ cnt, long rows)
{
    long i = (long)blockIdx.x * blockDim.x + threadIdx.x;
    const long st = (long)gridDim.x * blockDim.x;
    const long n = rows * OUTF;
    for (; i < n; i += st) out[i] /= fmaxf(cnt[i >> 8], 1.0f);
}

// ----------------------- LayerNorm (one wave / row) ------------------
__device__ __forceinline__ float wave_sum(float x)
{
#pragma unroll
    for (int o = 16; o > 0; o >>= 1) x += __shfl_xor(x, o, 32);
    return x;
}

__device__ __forceinline__ void ln8(float (&v)[8])
{
    float s = 0.0f;
#pragma unroll
    for (int j = 0; j < 8; ++j) s += v[j];
    s = wave_sum(s);
    const float mean = s * (1.0f / OUTF);
    float q = 0.0f;
#pragma unroll
    for (int j = 0; j < 8; ++j) { const float d = v[j] - mean; q += d * d; }
    q = wave_sum(q);
    const float inv = rsqrtf(q * (1.0f / OUTF) + 1e-5f);
#pragma unroll
    for (int j = 0; j < 8; ++j) v[j] = (v[j] - mean) * inv;
}

// out = act( LN(a (+b) (+c)) [*gamma + beta] )   — 8 waves per block
__global__ void ln_fuse_kernel(const float* __restrict__ a,
                               const float* __restrict__ b,
                               const float* __restrict__ c,
                               const float* __restrict__ gamma,
                               const float* __restrict__ beta,
                               float* __restrict__ out, long rows, int act)
{
    const int wave = threadIdx.x >> 5, lane = threadIdx.x & 31;
    const long row = (long)blockIdx.x * 8 + wave;
    if (row >= rows) return;
    float v[8];
#pragma unroll
    for (int j = 0; j < 8; ++j) {
        const long i = row * OUTF + j * 32 + lane;
        float x = a[i];
        if (b) x += b[i];
        if (c) x += c[i];
        v[j] = x;
    }
    ln8(v);
#pragma unroll
    for (int j = 0; j < 8; ++j) {
        const int f = j * 32 + lane;
        float y = v[j];
        if (gamma) y = y * gamma[f] + beta[f];
        if (act == ACT_RELU)     y = fmaxf(y, 0.0f);
        else if (act == ACT_ELU) y = (y > 0.0f) ? y : (__expf(y) - 1.0f);
        out[row * OUTF + f] = y;
    }
}

// ------------------------- final complex fuse ------------------------
__global__ void fuse_kernel(const float* __restrict__ x1,
                            const float* __restrict__ xfp,
                            const float* __restrict__ p2,
                            const float* __restrict__ xsk,
                            const int* __restrict__ idx1,
                            const int* __restrict__ idx2,
                            const float* __restrict__ wR,
                            const float* __restrict__ wI,
                            float* __restrict__ out, int npair)
{
    const int wave = threadIdx.x >> 5, lane = threadIdx.x & 31;
    const long i = (long)blockIdx.x * 8 + wave;
    if (i >= npair) return;
    const long a = idx1[i], b = idx2[i];
    float R1[8], I1[8], R2[8], I2[8];
#pragma unroll
    for (int j = 0; j < 8; ++j) {
        const int f = j * 32 + lane;
        R1[j] = x1 [a * OUTF + f] + xfp[b * OUTF + f];
        I1[j] = p2 [a * OUTF + f] + xsk[b * OUTF + f];
        R2[j] = xsk[a * OUTF + f] + p2 [b * OUTF + f];
        I2[j] = xfp[a * OUTF + f] + x1 [b * OUTF + f];
    }
    ln8(R1); ln8(I1); ln8(R2); ln8(I2);
    float R[8], I[8];
#pragma unroll
    for (int j = 0; j < 8; ++j) {
        R[j] = R1[j] * R2[j] - I1[j] * I2[j];
        I[j] = R1[j] * I2[j] + I1[j] * R2[j];
    }
    ln8(R); ln8(I);
    float acc = 0.0f;
#pragma unroll
    for (int j = 0; j < 8; ++j) {
        const int f = j * 32 + lane;
        // r_ + i_ = sum(R*wR + I*wI) + sum(I*wR - R*wI)
        acc += R[j] * (wR[f] - wI[f]) + I[j] * (wI[f] + wR[f]);
    }
    acc = wave_sum(acc);
    if (lane == 0) out[i] = 1.0f / (1.0f + __expf(-acc));
}

// ============================ host driver ============================
static inline void gemm(const float* A, const float* W, const float* bias,
                        float* C, int M, int N, int K, int act, hipStream_t s)
{
    const int tiles = (M / 32) * (N / 32);
    gemm_bf16_wmma<<<(tiles + 7) / 8, 256, 0, s>>>(A, W, bias, C, M, N, K, act);
}

static inline void gcn(const float* h, const int* ei, float* out,
                       const float* dinv, const float* bias,
                       int rows, int E, hipStream_t s)
{
    gcn_init_kernel<<<4096, 256, 0, s>>>(h, dinv, bias, out, rows);
    gcn_edge_kernel<<<E, 256, 0, s>>>(h, ei, ei + E, dinv, out, E);
}

static inline void smean(const float* t, const int* row_idx, const int* dst_idx,
                         const float* cnt, float* out, int rows, int E, hipStream_t s)
{
    fill_kernel<<<2048, 256, 0, s>>>(out, 0.0f, (long)rows * OUTF);
    smean_scatter_kernel<<<E, 256, 0, s>>>(t, row_idx, dst_idx, out, E);
    smean_div_kernel<<<4096, 256, 0, s>>>(out, cnt, rows);
}

static inline void lnf(const float* a, const float* b, const float* c,
                       const float* g, const float* bt,
                       float* out, long rows, int act, hipStream_t s)
{
    ln_fuse_kernel<<<(int)((rows + 7) / 8), 256, 0, s>>>(a, b, c, g, bt, out, rows, act);
}

extern "C" void kernel_launch(void* const* d_in, const int* in_sizes, int n_in,
                              void* d_out, int out_size, void* d_ws, size_t ws_size,
                              hipStream_t stream)
{
    (void)in_sizes; (void)n_in; (void)out_size; (void)ws_size;
    auto f = [&](int i) { return (const float*)d_in[i]; };

    const float* seq      = f(0);
    const int*   ppi      = (const int*)d_in[1];   // [2, EPPI]
    const int*   ddi      = (const int*)d_in[2];   // [2, EDDI]
    const int*   dti      = (const int*)d_in[3];   // [2, EDTI]
    const int*   idx1     = (const int*)d_in[4];
    const int*   idx2     = (const int*)d_in[5];
    const float* seq_init_W = f(6);
    const float* gs1_gcn_W = f(7),  *gs1_gcn_b = f(8),  *gs1_lin_W = f(9);
    const float* gs1_ln_w  = f(10), *gs1_ln_b  = f(11);
    const float* gs2_gcn_W = f(12), *gs2_gcn_b = f(13), *gs2_lin_W = f(14);
    const float* gs2_ln_w  = f(15), *gs2_ln_b  = f(16);
    const float* drug_emb  = f(17), *prot_emb  = f(18);
    const float* pp1_W = f(19), *pp1_b = f(20), *td1_W = f(21), *td1_b = f(22);
    const float* pr1_W = f(23), *pr1_b = f(24), *dd1_W = f(25), *dd1_b = f(26);
    const float* dt1_W = f(27), *dt1_b = f(28), *dr1_W = f(29), *dr1_b = f(30);
    const float* pp2_W = f(31), *pp2_b = f(32), *td2_W = f(33), *td2_b = f(34);
    const float* pr2_W = f(35), *pr2_b = f(36);
    // dd2/dt2/dr2 (37..42) are dead code in the ASC path
    const float* seq_fc_W = f(43), *skip_fc_W = f(44), *wR = f(45), *wI = f(46);

    // ---- workspace layout (12 big [20000,256] f32 buffers + small) ----
    const long NB = (long)Pn * OUTF;
    float* ws = (float*)d_ws;
    float* XSEQ = ws + 0 * NB;
    float* HBUF = ws + 1 * NB;
    float* X1   = ws + 2 * NB;
    float* DBUF = ws + 3 * NB;
    float* PBUF = ws + 4 * NB;
    float* P2B  = ws + 5 * NB;
    float* XFP  = ws + 6 * NB;
    float* XSK  = ws + 7 * NB;
    float* T1   = ws + 8 * NB;
    float* T2   = ws + 9 * NB;
    float* T3   = ws + 10 * NB;
    float* T4   = ws + 11 * NB;
    float* dinv_ppi = ws + 12 * NB;
    float* dinv_ddi = dinv_ppi + Pn;
    float* cnt_d    = dinv_ddi + Dn;
    float* cnt_p    = cnt_d + Dn;

    // ---- degrees / counts ----
    fill_kernel<<<256, 256, 0, stream>>>(dinv_ppi, 1.0f, Pn);   // self loops
    count_kernel<<<1024, 256, 0, stream>>>(ppi + EPPI, EPPI, dinv_ppi);
    rsqrt_kernel<<<(Pn + 255) / 256, 256, 0, stream>>>(dinv_ppi, Pn);
    fill_kernel<<<256, 256, 0, stream>>>(dinv_ddi, 1.0f, Dn);
    count_kernel<<<1024, 256, 0, stream>>>(ddi + EDDI, EDDI, dinv_ddi);
    rsqrt_kernel<<<(Dn + 255) / 256, 256, 0, stream>>>(dinv_ddi, Dn);
    fill_kernel<<<256, 256, 0, stream>>>(cnt_d, 0.0f, Dn);
    count_kernel<<<1024, 256, 0, stream>>>(dti, EDTI, cnt_d);          // dti[0]
    fill_kernel<<<256, 256, 0, stream>>>(cnt_p, 0.0f, Pn);
    count_kernel<<<1024, 256, 0, stream>>>(dti + EDTI, EDTI, cnt_p);   // dti[1]

    // ---- phase A: gnn_based_seq over PPI ----
    gemm(seq, seq_init_W, nullptr, XSEQ, Pn, OUTF, SEQL, ACT_RELU, stream);
    gemm(XSEQ, gs1_gcn_W, nullptr, T1, Pn, OUTF, OUTF, ACT_NONE, stream);
    gcn(T1, ppi, T2, dinv_ppi, gs1_gcn_b, Pn, EPPI, stream);
    gemm(XSEQ, gs1_lin_W, nullptr, T3, Pn, OUTF, OUTF, ACT_NONE, stream);
    lnf(T2, T3, nullptr, gs1_ln_w, gs1_ln_b, HBUF, Pn, ACT_RELU, stream);
    gemm(HBUF, gs2_gcn_W, nullptr, T1, Pn, OUTF, OUTF, ACT_NONE, stream);
    gcn(T1, ppi, T2, dinv_ppi, gs2_gcn_b, Pn, EPPI, stream);
    gemm(HBUF, gs2_lin_W, nullptr, T3, Pn, OUTF, OUTF, ACT_NONE, stream);
    lnf(T2, T3, nullptr, gs2_ln_w, gs2_ln_b, X1, Pn, ACT_NONE, stream);

    // ---- phase B: hetero layer 1 ----
    gemm(drug_emb, dd1_W, nullptr, T1, Dn, OUTF, OUTF, ACT_NONE, stream);
    gcn(T1, ddi, T2, dinv_ddi, dd1_b, Dn, EDDI, stream);
    gemm(prot_emb, dt1_W, dt1_b, T3, Pn, OUTF, OUTF, ACT_NONE, stream);
    smean(T3, dti + EDTI, dti, cnt_d, T4, Dn, EDTI, stream);           // prot->drug
    gemm(drug_emb, dr1_W, dr1_b, T1, Dn, OUTF, OUTF, ACT_NONE, stream);
    lnf(T2, T4, T1, nullptr, nullptr, DBUF, Dn, ACT_RELU, stream);

    gemm(prot_emb, pp1_W, nullptr, T1, Pn, OUTF, OUTF, ACT_NONE, stream);
    gcn(T1, ppi, T2, dinv_ppi, pp1_b, Pn, EPPI, stream);
    gemm(drug_emb, td1_W, td1_b, T3, Dn, OUTF, OUTF, ACT_NONE, stream);
    smean(T3, dti, dti + EDTI, cnt_p, T4, Pn, EDTI, stream);           // drug->prot
    gemm(prot_emb, pr1_W, pr1_b, T1, Pn, OUTF, OUTF, ACT_NONE, stream);
    lnf(T2, T4, T1, nullptr, nullptr, PBUF, Pn, ACT_RELU, stream);

    // ---- phase C: hetero layer 2 (protein update only; drug is dead) ----
    gemm(PBUF, pp2_W, nullptr, T1, Pn, OUTF, OUTF, ACT_NONE, stream);
    gcn(T1, ppi, T2, dinv_ppi, pp2_b, Pn, EPPI, stream);
    gemm(DBUF, td2_W, td2_b, T3, Dn, OUTF, OUTF, ACT_NONE, stream);
    smean(T3, dti, dti + EDTI, cnt_p, T4, Pn, EDTI, stream);
    gemm(PBUF, pr2_W, pr2_b, T1, Pn, OUTF, OUTF, ACT_NONE, stream);
    lnf(T2, T4, T1, nullptr, nullptr, P2B, Pn, ACT_NONE, stream);

    // ---- phase D: fuse inputs ----
    lnf(XSEQ, nullptr, nullptr, nullptr, nullptr, T1, Pn, ACT_ELU, stream); // fp
    gemm(T1, seq_fc_W, nullptr, XFP, Pn, OUTF, OUTF, ACT_ELU, stream);
    gemm(prot_emb, skip_fc_W, nullptr, XSK, Pn, OUTF, OUTF, ACT_NONE, stream);

    // ---- phase E: complex fuse -> sigmoid ----
    fuse_kernel<<<(NPAIR + 7) / 8, 256, 0, stream>>>(
        X1, XFP, P2B, XSK, idx1, idx2, wR, wI, (float*)d_out, NPAIR);
}
